// Block_88270167867464
// MI455X (gfx1250) — compile-verified, matrix operand order checked by reading, not run on hardware
//
#include <hip/hip_runtime.h>
#include <math.h>

// ---------------------------------------------------------------------------
// PVT-style block for MI455X (gfx1250, wave32, WMMA bf16 16x16x32).
// B=8, N=3136 (56x56), C=256, heads=4, dh=64, KV=784 (28x28), HID=1024.
// GEMMs: WMMA bf16 with async Global->LDS staging of the shared B panel.
// ---------------------------------------------------------------------------

typedef __attribute__((ext_vector_type(16))) __bf16 v16bf;
typedef __attribute__((ext_vector_type(8)))  __bf16 v8bf;
typedef __attribute__((ext_vector_type(8)))  float  v8f;

#define DEV __device__ __forceinline__

#define BB   8
#define NTOK 3136
#define CCH  256
#define HH   56
#define NH   4
#define DH   64
#define KVN  784
#define KVP  800      // KV padded to multiple of 32 for the attn@V K-loop
#define HID  1024
#define KC   64       // K-chunk staged per LDS buffer in k_gemm

DEV __bf16 f2bf(float f) {
  union { float f; unsigned u; } c; c.f = f;
  unsigned r = c.u + 0x7FFFu + ((c.u >> 16) & 1u);   // round-to-nearest-even
  unsigned short h = (unsigned short)(r >> 16);
  return __builtin_bit_cast(__bf16, h);
}
DEV float bf2f(__bf16 b) {
  unsigned short h = __builtin_bit_cast(unsigned short, b);
  union { unsigned u; float f; } c; c.u = ((unsigned)h) << 16;
  return c.f;
}
DEV float geluf(float x) { return 0.5f * x * (1.0f + erff(x * 0.70710678118654752f)); }

// 16-bit A/B fragment for V_WMMA_F32_16X16X32_BF16 from a row-major,
// K-contiguous matrix: lane (id&15) owns row/col, lanes 0-15 take K chunks
// {kk..kk+7, kk+16..kk+23}, lanes 16-31 take {kk+8.., kk+24..}.
DEV v16bf load_frag(const __bf16* base, int row, int ld, int kk, int sel) {
  const __bf16* p = base + (size_t)row * ld + kk + sel;
  v8bf lo = *(const v8bf*)p;
  v8bf hi = *(const v8bf*)(p + 16);
  v16bf r;
#pragma unroll
  for (int i = 0; i < 8; ++i) { r[i] = lo[i]; r[i + 8] = hi[i]; }
  return r;
}

DEV v8f wmma_bf16(v16bf a, v16bf b, v8f c) {
  return __builtin_amdgcn_wmma_f32_16x16x32_bf16(false, a, false, b, (short)0, c,
                                                 false, false);
}

// ---------------------------------------------------------------------------
// Small prep kernels
// ---------------------------------------------------------------------------
__global__ void k_bnprep(const float* g, const float* b, const float* m,
                         const float* v, float* s, float* t, int n) {
  int i = blockIdx.x * 256 + threadIdx.x;
  if (i < n) {
    float sc = g[i] * rsqrtf(v[i] + 1e-5f);
    s[i] = sc;
    t[i] = b[i] - m[i] * sc;
  }
}

__global__ void k_cvt(const float* in, __bf16* out, int n) {
  int i = blockIdx.x * 256 + threadIdx.x;
  if (i < n) out[i] = f2bf(in[i]);
}

__global__ void k_zerobf(__bf16* p, int n) {
  int i = blockIdx.x * 256 + threadIdx.x;
  if (i < n) p[i] = f2bf(0.f);
}

// ---------------------------------------------------------------------------
// LayerNorm over C=256, one wave per token (8 channels/lane, shfl reduce)
// ---------------------------------------------------------------------------
__global__ void __launch_bounds__(256) k_layernorm(const float* __restrict__ x,
                                                   const float* __restrict__ g,
                                                   const float* __restrict__ bsh,
                                                   __bf16* __restrict__ out,
                                                   int ntok) {
  int lane = threadIdx.x & 31;
  int wv = threadIdx.x >> 5;
  int tok = blockIdx.x * 8 + wv;
  if (tok >= ntok) return;
  const float* xp = x + (size_t)tok * CCH + lane * 8;
  float v[8];
  float s = 0.f, s2 = 0.f;
#pragma unroll
  for (int i = 0; i < 8; ++i) { v[i] = xp[i]; s += v[i]; s2 += v[i] * v[i]; }
#pragma unroll
  for (int m = 16; m >= 1; m >>= 1) {
    s  += __shfl_xor(s, m, 32);
    s2 += __shfl_xor(s2, m, 32);
  }
  float mean = s * (1.f / CCH);
  float var  = s2 * (1.f / CCH) - mean * mean;
  float rs   = rsqrtf(var + 1e-5f);
  __bf16* op = out + (size_t)tok * CCH + lane * 8;
#pragma unroll
  for (int i = 0; i < 8; ++i) {
    int c = lane * 8 + i;
    op[i] = f2bf((v[i] - mean) * rs * g[c] + bsh[c]);
  }
}

// ---------------------------------------------------------------------------
// Generic WMMA GEMM:  C[M,N] = A[M,K] * W[N,K]^T  (+ fused epilogues)
// Block = 8 waves, block tile 128M x 64N; each wave a 16x64 tile.
// The 64-col B panel (shared by all 8 waves) is staged Global->LDS with
// CDNA5 async-to-LDS DMA (double-buffered KC=64 chunks, s_wait_asynccnt),
// then consumed via ds_load_b128 fragments -> 8x less B vector traffic.
// mode 0: out bf16 = acc + bias
// mode 1: V-transpose store to vT[b][h][d][kv] (KVP stride, for attn@V)
// mode 2: out f32 = acc + bias + res              (attention out-proj)
// mode 3: out bf16 = BN(gelu(acc + bias))          (MLP fc1)
// mode 4: out f32 = BN(acc + bias) + res           (MLP fc2)
// ---------------------------------------------------------------------------
__global__ void __launch_bounds__(256) k_gemm(const __bf16* __restrict__ A,
                                              const __bf16* __restrict__ Bw,
                                              const float* __restrict__ bias,
                                              const float* __restrict__ res,
                                              const float* __restrict__ bs,
                                              const float* __restrict__ bt,
                                              float* __restrict__ outf,
                                              __bf16* __restrict__ outb,
                                              int M, int N, int K, int mode) {
  // double-buffered B panel: [buf][col 0..63][k 0..KC-1]
  __shared__ __bf16 panel[2][64 * KC];

  int lane = threadIdx.x & 31;
  int wv = threadIdx.x >> 5;
  int mt = blockIdx.y * 8 + wv;
  int nt = blockIdx.x;
  int sel = (lane & 16) ? 8 : 0;
  int colL = lane & 15;
  int mrow = mt * 16 + colL;

  // LDS byte offset of the panel: the generic (flat) address of __shared__
  // memory is {SHARED_BASE aperture, offset[31:0]}; HW truncates to [31:0],
  // so the low 32 bits are the wave-relative LDS offset needed by the async
  // DMA's VDST operand. The ptrtoint also makes `panel` escape so the asm's
  // memory clobber aliases it (keeps the ds_load fragment reads alive).
  unsigned ldsbase = (unsigned)(size_t)(void*)&panel[0][0];

  v8f acc[4] = {};

  // Stage one KC-wide chunk of the 64-col B panel into LDS buffer `bufi`.
  // 64 cols * KC * 2B = 8KB = 512 x 16B chunks; 256 threads x 2 async ops.
  auto stage = [&](int bufi, int kc) {
#pragma unroll
    for (int it = 0; it < 2; ++it) {
      int c = threadIdx.x + it * 256;                 // 0..511
      int col = c >> 3;                               // 0..63
      int o16 = c & 7;                                // 16B chunk in K dir
      const __bf16* g = Bw + (size_t)(nt * 64 + col) * K + kc + o16 * 8;
      unsigned lds = ldsbase + (unsigned)(bufi * (64 * KC * 2) +
                                          (col * KC + o16 * 8) * 2);
      asm volatile("global_load_async_to_lds_b128 %0, %1, off"
                   :: "v"(lds), "v"(g)
                   : "memory");
    }
  };

  int nchunks = K / KC;
  stage(0, 0);
  for (int ck = 0; ck < nchunks; ++ck) {
    int cur = ck & 1;
    if (ck + 1 < nchunks) {
      stage(cur ^ 1, (ck + 1) * KC);
      asm volatile("s_wait_asynccnt 2" ::: "memory");  // current buffer landed
    } else {
      asm volatile("s_wait_asynccnt 0" ::: "memory");
    }
    __syncthreads();

    const __bf16* pb = &panel[cur][0];
#pragma unroll
    for (int kkL = 0; kkL < KC; kkL += 32) {
      v16bf a = load_frag(A, mrow, K, ck * KC + kkL, sel);   // global A frag
#pragma unroll
      for (int f = 0; f < 4; ++f) {
        v16bf b = load_frag(pb, f * 16 + colL, KC, kkL, sel); // LDS B frag
        acc[f] = wmma_bf16(a, b, acc[f]);
      }
    }
    __syncthreads();   // all waves done reading before buffer is overwritten
  }

  int rb = mt * 16 + ((lane & 16) ? 8 : 0);
#pragma unroll
  for (int f = 0; f < 4; ++f) {
    int col = nt * 64 + f * 16 + colL;
    float bi = bias ? bias[col] : 0.f;
#pragma unroll
    for (int r = 0; r < 8; ++r) {
      int row = rb + r;
      size_t idx = (size_t)row * N + col;
      float val = acc[f][r] + bi;
      if (mode == 0) {
        outb[idx] = f2bf(val);
      } else if (mode == 1) {
        int b_ = row / KVN, j = row - b_ * KVN;
        int h = col >> 6, d = col & 63;
        outb[(((size_t)b_ * NH + h) * DH + d) * KVP + j] = f2bf(val);
      } else if (mode == 2) {
        outf[idx] = val + res[idx];
      } else if (mode == 3) {
        outb[idx] = f2bf(geluf(val) * bs[col] + bt[col]);
      } else {
        outf[idx] = val * bs[col] + bt[col] + res[idx];
      }
    }
  }
}

// ---------------------------------------------------------------------------
// Spatial-reduction: 2x2 stride-2 depthwise conv + BN on LN1 output (bf16)
// out: [B, KV, C] bf16, ready as K/V GEMM input
// ---------------------------------------------------------------------------
__global__ void k_srconv(const __bf16* __restrict__ hln, const float* __restrict__ w,
                         const float* __restrict__ bias, const float* __restrict__ s,
                         const float* __restrict__ t, __bf16* __restrict__ out) {
  int idx = blockIdx.x * 256 + threadIdx.x;           // BB*KVN*CCH
  int c = idx & (CCH - 1);
  int rest = idx >> 8;
  int n = rest % KVN;
  int b = rest / KVN;
  int i = n / 28, j = n % 28;
  float acc = bias[c];
#pragma unroll
  for (int di = 0; di < 2; ++di)
#pragma unroll
    for (int dj = 0; dj < 2; ++dj)
      acc += w[c * 4 + di * 2 + dj] *
             bf2f(hln[((size_t)b * NTOK + (2 * i + di) * HH + (2 * j + dj)) * CCH + c]);
  out[idx] = f2bf(acc * s[c] + t[c]);
}

// ---------------------------------------------------------------------------
// Attention: one workgroup per (b, head, 16-query tile).
// Phase 1: scores[16,KV] = (q kT)*scale + relpos via WMMA -> LDS (f32)
// Phase 2: rowwise softmax in LDS, probabilities -> bf16 LDS (KVP-padded)
// Phase 3: out[16,64] = P @ vT via WMMA (A from LDS via ds_load_b128)
// ---------------------------------------------------------------------------
__global__ void __launch_bounds__(256) k_attention(const __bf16* __restrict__ q,
                                                   const __bf16* __restrict__ kmat,
                                                   const __bf16* __restrict__ vT,
                                                   const float* __restrict__ relpos,
                                                   __bf16* __restrict__ attn_out) {
  extern __shared__ char smem[];
  float*  scores = (float*)smem;                               // 16*KVP f32
  __bf16* attn_s = (__bf16*)(smem + 16 * KVP * 4);             // 16*KVP bf16
  float*  rowmax = (float*)(smem + 16 * KVP * 4 + 16 * KVP * 2);
  float*  rowsum = rowmax + 16;

  int tid = threadIdx.x;
  int lane = tid & 31;
  int wv = tid >> 5;
  int colL = lane & 15;
  int sel = (lane & 16) ? 8 : 0;
  int rb  = (lane & 16) ? 8 : 0;
  int mt = blockIdx.x;
  int h  = blockIdx.y;
  int bi = blockIdx.z;

  const __bf16* qbase = q + ((size_t)bi * NTOK + mt * 16) * CCH + h * DH;
  v16bf a0 = load_frag(qbase, colL, CCH, 0, sel);
  v16bf a1 = load_frag(qbase, colL, CCH, 32, sel);
  const __bf16* kbase = kmat + (size_t)bi * KVN * CCH + h * DH;

  for (int jt = wv; jt < KVN / 16; jt += 8) {
    int col = jt * 16 + colL;
    v16bf b0 = load_frag(kbase, col, CCH, 0, sel);
    v16bf b1 = load_frag(kbase, col, CCH, 32, sel);
    v8f acc = {};
    acc = wmma_bf16(a0, b0, acc);
    acc = wmma_bf16(a1, b1, acc);
    const float* rp = relpos + ((size_t)h * NTOK + (size_t)mt * 16 + rb) * KVN + col;
#pragma unroll
    for (int r = 0; r < 8; ++r)
      scores[(rb + r) * KVP + col] = acc[r] * 0.125f + rp[(size_t)r * KVN];
  }
  __syncthreads();

  if (tid < 16) {
    float mx = -3.4e38f;
    for (int c = 0; c < KVN; ++c) mx = fmaxf(mx, scores[tid * KVP + c]);
    float sm = 0.f;
    for (int c = 0; c < KVN; ++c) sm += expf(scores[tid * KVP + c] - mx);
    rowmax[tid] = mx;
    rowsum[tid] = 1.f / sm;
  }
  __syncthreads();
  for (int idx = tid; idx < 16 * KVP; idx += 256) {
    int r = idx / KVP, c = idx - r * KVP;
    float p = (c < KVN) ? expf(scores[idx] - rowmax[r]) * rowsum[r] : 0.f;
    attn_s[idx] = f2bf(p);
  }
  __syncthreads();

  if (wv < 4) {                                   // 4 waves: one d-subtile each
    int d0 = wv * 16;
    const __bf16* vbase = vT + (((size_t)bi * NH + h) * DH + d0 + colL) * KVP;
    v8f acc = {};
    for (int kk = 0; kk < KVP; kk += 32) {
      v16bf a = load_frag(attn_s, colL, KVP, kk, sel);   // ds_load_b128 path
      const __bf16* p = vbase + kk + sel;
      v8bf lo = *(const v8bf*)p;
      v8bf hi = *(const v8bf*)(p + 16);
      v16bf b;
#pragma unroll
      for (int i = 0; i < 8; ++i) { b[i] = lo[i]; b[i + 8] = hi[i]; }
      acc = wmma_bf16(a, b, acc);
    }
    __bf16* op = attn_out + ((size_t)bi * NTOK + mt * 16 + rb) * CCH + h * DH + d0 + colL;
#pragma unroll
    for (int r = 0; r < 8; ++r) op[(size_t)r * CCH] = f2bf(acc[r]);
  }
}

// ---------------------------------------------------------------------------
// MLP middle: y2 = pBN(gelu(dwconv3x3(u) + dw_b + u)), u bf16 [B*N, HID]
// ---------------------------------------------------------------------------
__global__ void k_dwmlp(const __bf16* __restrict__ u, const float* __restrict__ w,
                        const float* __restrict__ bias, const float* __restrict__ ps,
                        const float* __restrict__ pt, __bf16* __restrict__ y2) {
  size_t idx = (size_t)blockIdx.x * 256 + threadIdx.x;   // BB*NTOK*HID
  int o = (int)(idx & (HID - 1));
  size_t rest = idx >> 10;
  int t = (int)(rest % NTOK);
  int b = (int)(rest / NTOK);
  int i = t / HH, j = t % HH;
  float acc = bias[o];
#pragma unroll
  for (int di = 0; di < 3; ++di) {
    int ii = i + di - 1;
    if (ii < 0 || ii >= HH) continue;
#pragma unroll
    for (int dj = 0; dj < 3; ++dj) {
      int jj = j + dj - 1;
      if (jj < 0 || jj >= HH) continue;
      acc += w[o * 9 + di * 3 + dj] *
             bf2f(u[((size_t)b * NTOK + ii * HH + jj) * HID + o]);
    }
  }
  float val = acc + bf2f(u[idx]);
  y2[idx] = f2bf(geluf(val) * ps[o] + pt[o]);
}

// ---------------------------------------------------------------------------
// Final depthwise 3x3 + bias + residual; writes f32 output [B, N, C]
// ---------------------------------------------------------------------------
__global__ void k_dwout(const float* __restrict__ x2, const float* __restrict__ w,
                        const float* __restrict__ bias, float* __restrict__ out) {
  size_t idx = (size_t)blockIdx.x * 256 + threadIdx.x;   // BB*NTOK*CCH
  int c = (int)(idx & (CCH - 1));
  size_t rest = idx >> 8;
  int t = (int)(rest % NTOK);
  int b = (int)(rest / NTOK);
  int i = t / HH, j = t % HH;
  float acc = bias[c];
#pragma unroll
  for (int di = 0; di < 3; ++di) {
    int ii = i + di - 1;
    if (ii < 0 || ii >= HH) continue;
#pragma unroll
    for (int dj = 0; dj < 3; ++dj) {
      int jj = j + dj - 1;
      if (jj < 0 || jj >= HH) continue;
      acc += w[c * 9 + di * 3 + dj] * x2[((size_t)b * NTOK + ii * HH + jj) * CCH + c];
    }
  }
  out[idx] = acc + x2[idx];
}

// ---------------------------------------------------------------------------
extern "C" void kernel_launch(void* const* d_in, const int* in_sizes, int n_in,
                              void* d_out, int out_size, void* d_ws, size_t ws_size,
                              hipStream_t stream) {
  const float* x      = (const float*)d_in[0];
  const float* relpos = (const float*)d_in[1];
  const float* ln1_g  = (const float*)d_in[2];
  const float* ln1_b  = (const float*)d_in[3];
  const float* wq     = (const float*)d_in[4];
  const float* bq     = (const float*)d_in[5];
  const float* wk     = (const float*)d_in[6];
  const float* bk     = (const float*)d_in[7];
  const float* wvw    = (const float*)d_in[8];
  const float* bv     = (const float*)d_in[9];
  const float* sr_w   = (const float*)d_in[10];
  const float* sr_b   = (const float*)d_in[11];
  const float* srbn_g = (const float*)d_in[12];
  const float* srbn_b = (const float*)d_in[13];
  const float* srbn_m = (const float*)d_in[14];
  const float* srbn_v = (const float*)d_in[15];
  const float* wo     = (const float*)d_in[16];
  const float* bo     = (const float*)d_in[17];
  const float* ln2_g  = (const float*)d_in[18];
  const float* ln2_b  = (const float*)d_in[19];
  const float* w1     = (const float*)d_in[20];
  const float* b1     = (const float*)d_in[21];
  const float* bn1_g  = (const float*)d_in[22];
  const float* bn1_b  = (const float*)d_in[23];
  const float* bn1_m  = (const float*)d_in[24];
  const float* bn1_v  = (const float*)d_in[25];
  const float* dw_w   = (const float*)d_in[26];
  const float* dw_b   = (const float*)d_in[27];
  const float* pbn_g  = (const float*)d_in[28];
  const float* pbn_b  = (const float*)d_in[29];
  const float* pbn_m  = (const float*)d_in[30];
  const float* pbn_v  = (const float*)d_in[31];
  const float* w2     = (const float*)d_in[32];
  const float* b2     = (const float*)d_in[33];
  const float* bn2_g  = (const float*)d_in[34];
  const float* bn2_b  = (const float*)d_in[35];
  const float* bn2_m  = (const float*)d_in[36];
  const float* bn2_v  = (const float*)d_in[37];
  const float* bdw_w  = (const float*)d_in[38];
  const float* bdw_b  = (const float*)d_in[39];

  const size_t MTOK = (size_t)BB * NTOK;        // 25088
  const size_t MKV  = (size_t)BB * KVN;         // 6272

  char* base = (char*)d_ws;
  size_t off = 0;
  auto carve = [&](size_t bytes) -> void* {
    void* p = base + off;
    off = (off + bytes + 255) & ~(size_t)255;
    return p;
  };

  __bf16* hln  = (__bf16*)carve(MTOK * CCH * 2);
  __bf16* qb   = (__bf16*)carve(MTOK * CCH * 2);
  __bf16* xsr  = (__bf16*)carve(MKV * CCH * 2);
  __bf16* kb   = (__bf16*)carve(MKV * CCH * 2);
  __bf16* vT   = (__bf16*)carve((size_t)BB * NH * DH * KVP * 2);
  __bf16* aout = (__bf16*)carve(MTOK * CCH * 2);
  float*  xat  = (float*)carve(MTOK * CCH * 4);
  __bf16* h2   = (__bf16*)carve(MTOK * CCH * 2);
  __bf16* ub   = (__bf16*)carve(MTOK * HID * 2);
  __bf16* y2   = (__bf16*)carve(MTOK * HID * 2);
  float*  x2   = (float*)carve(MTOK * CCH * 4);
  __bf16* wqb  = (__bf16*)carve((size_t)CCH * CCH * 2);
  __bf16* wkb  = (__bf16*)carve((size_t)CCH * CCH * 2);
  __bf16* wvb  = (__bf16*)carve((size_t)CCH * CCH * 2);
  __bf16* wob  = (__bf16*)carve((size_t)CCH * CCH * 2);
  __bf16* w1b  = (__bf16*)carve((size_t)HID * CCH * 2);
  __bf16* w2b  = (__bf16*)carve((size_t)CCH * HID * 2);
  float* srS = (float*)carve(CCH * 4);
  float* srT = (float*)carve(CCH * 4);
  float* b1S = (float*)carve(HID * 4);
  float* b1T = (float*)carve(HID * 4);
  float* pS  = (float*)carve(HID * 4);
  float* pT  = (float*)carve(HID * 4);
  float* b2S = (float*)carve(CCH * 4);
  float* b2T = (float*)carve(CCH * 4);

  // --- prep: BN scale/shift, weights -> bf16, zero padded vT -------------
  k_bnprep<<<1, 256, 0, stream>>>(srbn_g, srbn_b, srbn_m, srbn_v, srS, srT, CCH);
  k_bnprep<<<4, 256, 0, stream>>>(bn1_g, bn1_b, bn1_m, bn1_v, b1S, b1T, HID);
  k_bnprep<<<4, 256, 0, stream>>>(pbn_g, pbn_b, pbn_m, pbn_v, pS, pT, HID);
  k_bnprep<<<1, 256, 0, stream>>>(bn2_g, bn2_b, bn2_m, bn2_v, b2S, b2T, CCH);
  k_cvt<<<256, 256, 0, stream>>>(wq, wqb, CCH * CCH);
  k_cvt<<<256, 256, 0, stream>>>(wk, wkb, CCH * CCH);
  k_cvt<<<256, 256, 0, stream>>>(wvw, wvb, CCH * CCH);
  k_cvt<<<256, 256, 0, stream>>>(wo, wob, CCH * CCH);
  k_cvt<<<1024, 256, 0, stream>>>(w1, w1b, HID * CCH);
  k_cvt<<<1024, 256, 0, stream>>>(w2, w2b, CCH * HID);
  k_zerobf<<<(BB * NH * DH * KVP) / 256, 256, 0, stream>>>(vT, BB * NH * DH * KVP);

  // --- attention ---------------------------------------------------------
  k_layernorm<<<MTOK / 8, 256, 0, stream>>>(x, ln1_g, ln1_b, hln, (int)MTOK);
  k_gemm<<<dim3(CCH / 64, MTOK / 128), 256, 0, stream>>>(
      hln, wqb, bq, nullptr, nullptr, nullptr, nullptr, qb,
      (int)MTOK, CCH, CCH, 0);
  k_srconv<<<(int)(MKV * CCH / 256), 256, 0, stream>>>(hln, sr_w, sr_b, srS, srT, xsr);
  k_gemm<<<dim3(CCH / 64, MKV / 128), 256, 0, stream>>>(
      xsr, wkb, bk, nullptr, nullptr, nullptr, nullptr, kb,
      (int)MKV, CCH, CCH, 0);
  k_gemm<<<dim3(CCH / 64, MKV / 128), 256, 0, stream>>>(
      xsr, wvb, bv, nullptr, nullptr, nullptr, nullptr, vT,
      (int)MKV, CCH, CCH, 1);

  size_t smem_bytes = (size_t)16 * KVP * 4 + (size_t)16 * KVP * 2 + 128;
  k_attention<<<dim3(NTOK / 16, NH, BB), 256, smem_bytes, stream>>>(
      qb, kb, vT, relpos, aout);

  k_gemm<<<dim3(CCH / 64, MTOK / 128), 256, 0, stream>>>(
      aout, wob, bo, x, nullptr, nullptr, xat, nullptr,
      (int)MTOK, CCH, CCH, 2);

  // --- conv MLP ----------------------------------------------------------
  k_layernorm<<<MTOK / 8, 256, 0, stream>>>(xat, ln2_g, ln2_b, h2, (int)MTOK);
  k_gemm<<<dim3(HID / 64, MTOK / 128), 256, 0, stream>>>(
      h2, w1b, b1, nullptr, b1S, b1T, nullptr, ub,
      (int)MTOK, HID, CCH, 3);
  k_dwmlp<<<(int)(MTOK * HID / 256), 256, 0, stream>>>(ub, dw_w, dw_b, pS, pT, y2);
  k_gemm<<<dim3(CCH / 64, MTOK / 128), 256, 0, stream>>>(
      y2, w2b, b2, xat, b2S, b2T, x2, nullptr,
      (int)MTOK, CCH, HID, 4);

  // --- final depthwise residual ------------------------------------------
  k_dwout<<<(int)(MTOK * CCH / 256), 256, 0, stream>>>(x2, bdw_w, bdw_b, (float*)d_out);
}
